// MultiheadLinearSelfAttentionKernalizationLayer_43731357008677
// MI455X (gfx1250) — matrix operand order repaired and checked by reading, not compiled, not run on hardware
//
#include <hip/hip_runtime.h>
#include <math.h>

typedef _Float16 h16;
typedef __attribute__((ext_vector_type(8)))  _Float16 v8h;
typedef __attribute__((ext_vector_type(16))) _Float16 v16h;
typedef __attribute__((ext_vector_type(8)))  float    v8f;

constexpr int  BB = 64, SS = 256, DD = 2048, HH = 8, DHH = 256;
constexpr float EPSF = 1e-6f;

__device__ __forceinline__ float elu1f(float x) { return x > 0.0f ? x + 1.0f : __expf(x); }

// ---------------------------------------------------------------------------
// Generic batched WMMA GEMM:  C[b] = A[b] (MxK, row-major) @ Bt[b]^T  (+ epilogue)
// Bt is stored [N][K] row-major so both A and B fragments are contiguous in K.
// Block tile 128x128x64, 256 threads = 8 wave32 waves (4(m) x 2(n)),
// each wave owns a 32x64 tile = 2x4 accumulators of v_wmma_f32_16x16x32_f16,
// two K=32 sub-steps per tile -> 16 WMMAs per wave per barrier.
// Global->LDS staging uses GLOBAL_LOAD_ASYNC_TO_LDS_B128 (ASYNCcnt) with
// double-buffered LDS so the next tile's DMA overlaps the current tile's math.
// ---------------------------------------------------------------------------
struct GemmParams {
  const h16* A;  long lda; long sA0, sA1;
  const h16* Bt; long ldb; long sB0, sB1;
  long M, N, K;
  int  batchDiv;                       // bz -> (bz/div, bz%div) two-level batch offsets
  float* Cf; h16* Ch; long ldc; long sC0, sC1;
  const float* bias;                   // [N] or null
  const float* scale; long sS0, sS1;   // per-column scale (EPI_SCALE)
  const float* addF;                   // residual (EPI_O)
  const float* sumo;                   // [B, D] summary contribution (EPI_O)
};

enum { EPI_Q = 0, EPI_ELU, EPI_VT, EPI_SCALE, EPI_F16, EPI_F32, EPI_O };

__device__ __forceinline__ void async_b128(unsigned lds_addr, unsigned long long gaddr) {
  asm volatile("global_load_async_to_lds_b128 %0, %1, off"
               :: "v"(lds_addr), "v"(gaddr) : "memory");
}
__device__ __forceinline__ void wait_async0() {
  asm volatile("s_wait_asynccnt 0" ::: "memory");
}
__device__ __forceinline__ unsigned lds_off(const void* p) {
  return (unsigned)(unsigned long long)p;   // flat LDS addr: low 32 bits = LDS offset
}

template <int EPI>
__global__ __launch_bounds__(256) void gemm_wmma(GemmParams p) {
  constexpr int BK  = 64;
  constexpr int LDT = 72;              // 64 + 8 halves pad: 144B rows -> conflict-free b128
  __shared__ __align__(16) h16 Asm[2][128 * LDT];
  __shared__ __align__(16) h16 Bsm[2][128 * LDT];

  const int  bz = blockIdx.z;
  const long bo_ = (long)(bz / p.batchDiv);
  const long bi_ = (long)(bz % p.batchDiv);
  const long aoff = bo_ * p.sA0 + bi_ * p.sA1;
  const long boff = bo_ * p.sB0 + bi_ * p.sB1;
  const long coff = bo_ * p.sC0 + bi_ * p.sC1;

  const int tid = threadIdx.x, lane = tid & 31, wid = tid >> 5;
  const int wm = wid >> 1, wn = wid & 1;

  const long m0 = (long)blockIdx.y * 128;
  const long n0 = (long)blockIdx.x * 128;

  const h16* Ab = p.A + aoff;
  const h16* Bb = p.Bt + boff;

  // Issue one 128xBK tile (A and Bt) as async LDS copies: 8 x b128 per thread.
  auto issue_tile = [&](int buf, long kk) {
#pragma unroll
    for (int c = 0; c < 4; ++c) {
      int idx = tid + c * 256;          // 0..1023 -> 8-half chunk id
      int row = idx >> 3;
      int col = (idx & 7) << 3;
      long gm = m0 + row; if (gm >= p.M) gm = p.M - 1;   // M-guard (clamped rows unused)
      async_b128(lds_off(&Asm[buf][row * LDT + col]),
                 (unsigned long long)(Ab + gm * p.lda + kk + col));
      long gn = n0 + row;                                 // N always multiple of 128 here
      async_b128(lds_off(&Bsm[buf][row * LDT + col]),
                 (unsigned long long)(Bb + gn * p.ldb + kk + col));
    }
  };

  v8f zf = {};
  v8f acc[2][4];
#pragma unroll
  for (int i = 0; i < 2; ++i)
#pragma unroll
    for (int j = 0; j < 4; ++j) acc[i][j] = zf;

  const long ntiles = p.K >> 6;
  issue_tile(0, 0);

  const int fr    = lane & 15;
  const int ahalf = (lane >> 4) << 3;   // A: K-base 0 / 8 within 32-substep, 2nd chunk +16
  const int bhalf = (lane >> 4) << 3;   // B: K-base 0 / 8 within 32-substep, 2nd chunk +16

  for (long t = 0; t < ntiles; ++t) {
    const int cur = (int)(t & 1);
    wait_async0();                      // own copies for tile t landed in LDS
    __syncthreads();                    // everyone's copies landed; buf[cur^1] free
    if (t + 1 < ntiles) issue_tile(cur ^ 1, (t + 1) << 6);   // overlap with compute

#pragma unroll
    for (int ks = 0; ks < BK; ks += 32) {
      v16h af[2], bf[4];
#pragma unroll
      for (int mi = 0; mi < 2; ++mi) {
        const h16* ap = &Asm[cur][(wm * 32 + mi * 16 + fr) * LDT + ks + ahalf];
        v8h lo = *(const v8h*)ap;
        v8h hi = *(const v8h*)(ap + 16);
        af[mi] = __builtin_shufflevector(lo, hi, 0,1,2,3,4,5,6,7,8,9,10,11,12,13,14,15);
      }
#pragma unroll
      for (int ni = 0; ni < 4; ++ni) {
        const h16* bp = &Bsm[cur][(wn * 64 + ni * 16 + fr) * LDT + ks + bhalf];
        v8h lo = *(const v8h*)bp;
        v8h hi = *(const v8h*)(bp + 16);
        bf[ni] = __builtin_shufflevector(lo, hi, 0,1,2,3,4,5,6,7,8,9,10,11,12,13,14,15);
      }
#pragma unroll
      for (int mi = 0; mi < 2; ++mi)
#pragma unroll
        for (int ni = 0; ni < 4; ++ni)
          acc[mi][ni] = __builtin_amdgcn_wmma_f32_16x16x32_f16(
              false, af[mi], false, bf[ni], (short)0, acc[mi][ni], false, false);
    }
    __syncthreads();                    // done reading buf[cur] before t+1 overwrites it
  }

  // Epilogue: D element v of lane -> (M = 8*(lane/16)+v, N = lane%16) in 16x16 tile.
  const int cl = lane & 15;
  const int rb = (lane >> 4) << 3;
#pragma unroll
  for (int mi = 0; mi < 2; ++mi) {
#pragma unroll
    for (int ni = 0; ni < 4; ++ni) {
      long gn = n0 + wn * 64 + ni * 16 + cl;
      float bvv = p.bias ? p.bias[gn] : 0.0f;
      float sc = 1.0f;
      if (EPI == EPI_SCALE) sc = p.scale[bo_ * p.sS0 + bi_ * p.sS1 + gn];
#pragma unroll
      for (int v = 0; v < 8; ++v) {
        long gm = m0 + wm * 32 + mi * 16 + rb + v;
        if (gm >= p.M) continue;
        float val = acc[mi][ni][v] + bvv;
        long ci = coff + gm * p.ldc + gn;
        if (EPI == EPI_Q)        { p.Cf[ci] = val; p.Ch[ci] = (h16)elu1f(val); }
        else if (EPI == EPI_ELU) { p.Ch[ci] = (h16)elu1f(val); }
        else if (EPI == EPI_VT)  {          // store V transposed: [B,H,DH,S]
          long b = gm / SS, s = gm % SS, h = gn / DHH, d = gn % DHH;
          p.Ch[((b * HH + h) * (long)DHH + d) * SS + s] = (h16)val;
        }
        else if (EPI == EPI_SCALE) { p.Ch[ci] = (h16)(val * sc); }
        else if (EPI == EPI_F16)   { p.Ch[ci] = (h16)val; }
        else if (EPI == EPI_F32)   { p.Cf[ci] = val; }
        else if (EPI == EPI_O)     {        // z = attn@Wo_top + bo + q_proj + summary@Wo_bot
          long b = gm / SS;
          p.Cf[ci] = val + p.addF[ci] + p.sumo[b * DD + gn];
        }
      }
    }
  }
}

// --------------------------- support kernels --------------------------------

__global__ __launch_bounds__(256) void cvt_f16(const float* __restrict__ in,
                                               h16* __restrict__ out, long n) {
  long i = (long)blockIdx.x * blockDim.x + threadIdx.x;
  long stride = (long)gridDim.x * blockDim.x;
  for (; i < n; i += stride) out[i] = (h16)in[i];
}

// W [K][N] f32 -> Wt [N][K] f16 (coalesced both ways via LDS tile)
__global__ __launch_bounds__(256) void transpose_cvt(const float* __restrict__ W,
                                                     h16* __restrict__ Wt, int K, int N) {
  __shared__ float t[32][33];
  int n0 = blockIdx.x * 32, k0 = blockIdx.y * 32;
#pragma unroll
  for (int i = 0; i < 4; ++i)
    t[threadIdx.y + 8 * i][threadIdx.x] =
        W[(long)(k0 + threadIdx.y + 8 * i) * N + n0 + threadIdx.x];
  __syncthreads();
#pragma unroll
  for (int i = 0; i < 4; ++i)
    Wt[(long)(n0 + threadIdx.y + 8 * i) * K + k0 + threadIdx.x] =
        (h16)t[threadIdx.x][threadIdx.y + 8 * i];
}

// winv[b*D + c] = 1 / (sum_s kh[b,s,c] + eps)   (c = h*DH + j, matches score col j)
__global__ __launch_bounds__(256) void keysum_inv(const h16* __restrict__ kh,
                                                  float* __restrict__ winv) {
  long i = (long)blockIdx.x * blockDim.x + threadIdx.x;   // B*D threads
  long b = i / DD, c = i % DD;
  const h16* p = kh + b * (long)SS * DD + c;
  float s = 0.0f;
  for (int j = 0; j < SS; ++j) s += (float)p[(long)j * DD];
  winv[i] = 1.0f / (s + EPSF);
}

// colsum[b*D + c] = (1/S) * sum_s attn[b,s,c]   (f16 out, feeds summary GEMM)
__global__ __launch_bounds__(256) void colsum_mean(const h16* __restrict__ attn,
                                                   h16* __restrict__ cs) {
  long i = (long)blockIdx.x * blockDim.x + threadIdx.x;   // B*D threads
  long b = i / DD, c = i % DD;
  const h16* p = attn + b * (long)SS * DD + c;
  float s = 0.0f;
  for (int j = 0; j < SS; ++j) s += (float)p[(long)j * DD];
  cs[i] = (h16)(s * (1.0f / SS));
}

// Row-wise LayerNorm over D=2048, one 256-thread block per row.
__global__ __launch_bounds__(256) void layernorm_k(const float* __restrict__ z,
                                                   const float* __restrict__ gamma,
                                                   const float* __restrict__ beta,
                                                   float* __restrict__ out) {
  const long r = blockIdx.x;
  const float* zr = z + r * (long)DD;
  const int tid = threadIdx.x;
  float s = 0.0f, s2 = 0.0f;
  for (int c = tid; c < DD; c += 256) { float v = zr[c]; s += v; s2 += v * v; }
  __shared__ float rs[256], rs2[256];
  rs[tid] = s; rs2[tid] = s2;
  __syncthreads();
  for (int off = 128; off > 0; off >>= 1) {
    if (tid < off) { rs[tid] += rs[tid + off]; rs2[tid] += rs2[tid + off]; }
    __syncthreads();
  }
  float mu = rs[0] * (1.0f / DD);
  float var = rs2[0] * (1.0f / DD) - mu * mu;
  float rstd = rsqrtf(var + EPSF);
  for (int c = tid; c < DD; c += 256)
    out[r * (long)DD + c] = (zr[c] - mu) * rstd * gamma[c] + beta[c];
}

// ---------------------------------------------------------------------------

extern "C" void kernel_launch(void* const* d_in, const int* in_sizes, int n_in,
                              void* d_out, int out_size, void* d_ws, size_t ws_size,
                              hipStream_t stream) {
  const float* x  = (const float*)d_in[0];
  const float* Wq = (const float*)d_in[1];
  const float* bq = (const float*)d_in[2];
  const float* Wk = (const float*)d_in[3];
  const float* bk = (const float*)d_in[4];
  const float* Wv = (const float*)d_in[5];
  const float* bv = (const float*)d_in[6];
  const float* Wo = (const float*)d_in[7];
  const float* bo = (const float*)d_in[8];
  const float* gamma = (const float*)d_in[9];
  const float* beta  = (const float*)d_in[10];
  float* out = (float*)d_out;

  const size_t MB = (size_t)1 << 20;
  char* ws = (char*)d_ws;
  // Workspace layout (aggressively aliased; ~394 MB peak):
  h16*   xh     = (h16*)(ws + 0);             //  64 MB, dead after V-GEMM
  h16*   scores = (h16*)(ws + 0);             //  64 MB, aliases xh
  float* qproj  = (float*)(ws + 64 * MB);     // 128 MB, becomes z in-place
  h16*   qh     = (h16*)(ws + 192 * MB);      //  64 MB, dead after S-GEMM
  h16*   attn   = (h16*)(ws + 192 * MB);      //  64 MB, aliases qh
  h16*   kh     = (h16*)(ws + 256 * MB);      //  64 MB
  h16*   vT     = (h16*)(ws + 320 * MB);      //  64 MB
  h16*   wbuf   = (h16*)(ws + 384 * MB);      //   8 MB, shared by all 5 weight xposes
  float* winv   = (float*)(ws + 392 * MB);                    // 512 KB
  h16*   csum   = (h16*)(ws + 392 * MB + 512 * 1024);         // 256 KB
  float* sumo   = (float*)(ws + 392 * MB + 768 * 1024);       // 512 KB
  if (ws_size < 394 * MB) return;

  const long NROW = (long)BB * SS;            // 16384
  dim3 tblk(32, 8);
  dim3 tgrd(DD / 32, DD / 32);

  // 1) x -> f16
  cvt_f16<<<4096, 256, 0, stream>>>(x, xh, NROW * DD);

  GemmParams g{};
  g.batchDiv = 1;

  // 2) Q-GEMM: q_proj (f32) + qh = elu(q_proj)+1 (f16)
  transpose_cvt<<<tgrd, tblk, 0, stream>>>(Wq, wbuf, DD, DD);
  g.A = xh;  g.lda = DD; g.sA0 = g.sA1 = 0;
  g.Bt = wbuf; g.ldb = DD; g.sB0 = g.sB1 = 0;
  g.M = NROW; g.N = DD; g.K = DD;
  g.Cf = qproj; g.Ch = qh; g.ldc = DD; g.sC0 = g.sC1 = 0;
  g.bias = bq;
  gemm_wmma<EPI_Q><<<dim3(DD / 128, NROW / 128, 1), 256, 0, stream>>>(g);

  // 3) K-GEMM: kh = elu(x@Wk+bk)+1
  transpose_cvt<<<tgrd, tblk, 0, stream>>>(Wk, wbuf, DD, DD);
  g.Cf = nullptr; g.Ch = kh; g.bias = bk;
  gemm_wmma<EPI_ELU><<<dim3(DD / 128, NROW / 128, 1), 256, 0, stream>>>(g);

  // 4) V-GEMM: vT[b,h,d,s] = (x@Wv+bv)[b,s,h*DH+d]
  transpose_cvt<<<tgrd, tblk, 0, stream>>>(Wv, wbuf, DD, DD);
  g.Ch = vT; g.bias = bv;
  gemm_wmma<EPI_VT><<<dim3(DD / 128, NROW / 128, 1), 256, 0, stream>>>(g);

  // 5) winv = 1/(key_sum + eps)
  keysum_inv<<<(BB * DD) / 256, 256, 0, stream>>>(kh, winv);

  // 6) scores-GEMM (batched over B*H): scores = (q @ k^T) * winv[col]
  GemmParams gs{};
  gs.batchDiv = HH;
  gs.A = qh; gs.lda = DD; gs.sA0 = (long)SS * DD; gs.sA1 = DHH;
  gs.Bt = kh; gs.ldb = DD; gs.sB0 = (long)SS * DD; gs.sB1 = DHH;
  gs.M = SS; gs.N = SS; gs.K = DHH;
  gs.Ch = scores; gs.ldc = SS; gs.sC0 = (long)HH * SS * SS; gs.sC1 = (long)SS * SS;
  gs.scale = winv; gs.sS0 = DD; gs.sS1 = DHH;
  gemm_wmma<EPI_SCALE><<<dim3(SS / 128, SS / 128, BB * HH), 256, 0, stream>>>(gs);

  // 7) attn-GEMM (batched): attn[b,s,h*DH+d] = scores @ v
  GemmParams ga{};
  ga.batchDiv = HH;
  ga.A = scores; ga.lda = SS; ga.sA0 = (long)HH * SS * SS; ga.sA1 = (long)SS * SS;
  ga.Bt = vT; ga.ldb = SS; ga.sB0 = (long)HH * DHH * SS; ga.sB1 = (long)DHH * SS;
  ga.M = SS; ga.N = DHH; ga.K = SS;
  ga.Ch = attn; ga.ldc = DD; ga.sC0 = (long)SS * DD; ga.sC1 = DHH;
  gemm_wmma<EPI_F16><<<dim3(DHH / 128, SS / 128, BB * HH), 256, 0, stream>>>(ga);

  // 8) summary path: colsum -> sumo = mean(attn) @ Wo_bot  (M=64, M-guarded)
  colsum_mean<<<(BB * DD) / 256, 256, 0, stream>>>(attn, csum);
  transpose_cvt<<<tgrd, tblk, 0, stream>>>(Wo + (long)DD * DD, wbuf, DD, DD);
  GemmParams gm{};
  gm.batchDiv = 1;
  gm.A = csum; gm.lda = DD;
  gm.Bt = wbuf; gm.ldb = DD;
  gm.M = BB; gm.N = DD; gm.K = DD;
  gm.Cf = sumo; gm.ldc = DD;
  gemm_wmma<EPI_F32><<<dim3(DD / 128, 1, 1), 256, 0, stream>>>(gm);

  // 9) O-GEMM: z = attn@Wo_top + bo + q_proj + sumo[b]   (written over qproj)
  transpose_cvt<<<tgrd, tblk, 0, stream>>>(Wo, wbuf, DD, DD);
  GemmParams go{};
  go.batchDiv = 1;
  go.A = attn; go.lda = DD;
  go.Bt = wbuf; go.ldb = DD;
  go.M = NROW; go.N = DD; go.K = DD;
  go.Cf = qproj; go.ldc = DD;
  go.bias = bo; go.addF = qproj; go.sumo = sumo;
  gemm_wmma<EPI_O><<<dim3(DD / 128, NROW / 128, 1), 256, 0, stream>>>(go);

  // 10) LayerNorm -> out
  layernorm_k<<<NROW, 256, 0, stream>>>(qproj, gamma, beta, out);

  (void)in_sizes; (void)n_in; (void)out_size;
}
